// LongCatSelfAttention_6476810682931
// MI455X (gfx1250) — matrix operand order repaired and checked by reading, not compiled, not run on hardware
//
#include <hip/hip_runtime.h>
#include <hip/hip_bf16.h>

typedef __attribute__((ext_vector_type(16))) _Float16 v16h;
typedef __attribute__((ext_vector_type(8)))  _Float16 v8h;
typedef __attribute__((ext_vector_type(8)))  float    v8f;

#define DIMC   2048
#define NTOK   4096
#define NHEADS 16
#define HDIM   128

union HF16 { v16h v; v8h h[2]; _Float16 e[16]; };

// ---- CDNA5 Tensor Data Mover (guarded: falls back to cooperative loads) ----
#if defined(__has_builtin)
#if __has_builtin(__builtin_amdgcn_tensor_load_to_lds)
#define HAVE_TDM 1
#endif
#endif
#ifndef HAVE_TDM
#define HAVE_TDM 0
#endif

#if defined(__has_builtin)
#if __has_builtin(__builtin_amdgcn_s_wait_tensorcnt)
#define WAIT_TENSOR(n) __builtin_amdgcn_s_wait_tensorcnt((short)(n))
#endif
#endif
#ifndef WAIT_TENSOR
#define WAIT_TENSOR(n) asm volatile("s_wait_tensorcnt %0" :: "i"(n) : "memory")
#endif

#if HAVE_TDM
typedef __attribute__((ext_vector_type(4))) unsigned int tdm_u32x4;
typedef __attribute__((ext_vector_type(8))) int          tdm_i32x8;
typedef __attribute__((ext_vector_type(4))) int          tdm_i32x4;

// DMA one [32 x 128] f16 tile (rows stride DIMC) from global -> LDS (row-major).
// D# per CDNA5 ISA.
static __device__ __forceinline__ void tdm_load_tile_f16(const _Float16* gsrc,
                                                         unsigned int ldsOff) {
    unsigned long long ga = (unsigned long long)(uintptr_t)gsrc;
    tdm_u32x4 g0;
    g0[0] = 1u;                                              // count=1, user mode
    g0[1] = ldsOff;                                          // lds_addr (bytes)
    g0[2] = (unsigned int)ga;                                // global_addr[31:0]
    g0[3] = (unsigned int)((ga >> 32) & 0x01FFFFFFu)         // global_addr[56:32]
          | (2u << 30);                                      // type = 2 ("image")
    tdm_i32x8 g1;
    g1[0] = 1 << 16;                                         // data_size=1 -> 2 bytes
    g1[1] = (int)(((unsigned)DIMC & 0xFFFFu) << 16);         // tensor_dim0 lo16 (2048)
    g1[2] = (int)((((unsigned)DIMC) >> 16)
          | (((unsigned)NTOK & 0xFFFFu) << 16));             // dim0 hi16, dim1 lo16 (4096)
    g1[3] = (int)((((unsigned)NTOK) >> 16) | (128u << 16));  // dim1 hi16, tile_dim0=128
    g1[4] = 32;                                              // tile_dim1=32, tile_dim2=0
    g1[5] = DIMC;                                            // tensor_dim0_stride = 2048
    g1[6] = 0;
    g1[7] = 0;
    tdm_i32x4 gz4 = {0, 0, 0, 0};
    tdm_i32x8 gz8 = {0, 0, 0, 0, 0, 0, 0, 0};
    // 6-arg form (clang-23 / therock-10.0): (g0, g1, g2, g3, g_extra, cpol)
    __builtin_amdgcn_tensor_load_to_lds(g0, g1, gz4, gz4, gz8, 0);
}
#endif

// DPP cross-lane combine within a 16-lane row (full-rate VALU, no LDS)
#define DPPF(x, ctrl) __builtin_bit_cast(float, __builtin_amdgcn_update_dpp( \
    0, __builtin_bit_cast(int, (x)), (ctrl), 0xF, 0xF, true))

// max across the 16-lane group: xor1, xor2 via quad_perm; 4s->8s->16 via mirrors
static __device__ __forceinline__ float rowmax16(float x) {
    x = fmaxf(x, DPPF(x, 0xB1));   // quad_perm [1,0,3,2]
    x = fmaxf(x, DPPF(x, 0x4E));   // quad_perm [2,3,0,1]
    x = fmaxf(x, DPPF(x, 0x141));  // row_half_mirror
    x = fmaxf(x, DPPF(x, 0x140));  // row_mirror
    return x;
}

// A-matrix 16x32 f16 fragment (row-major source, leading dim ld):
// lane L holds row M=L&15; halves [kc, kc+8) and [kc+16, kc+24), kc=(L>>4)*8
static __device__ __forceinline__ v16h load_a_frag(const _Float16* base, int lane, int ld) {
    int m  = lane & 15;
    int kc = (lane >> 4) * 8;
    const _Float16* p = base + (size_t)m * ld + kc;
    HF16 a;
    a.h[0] = *(const v8h*)(p);
    a.h[1] = *(const v8h*)(p + 16);
    return a.v;
}

// B-matrix 32x16 f16 fragment where column n of B == row n of `base`
// lane L: n=L&15; halves [kc, kc+8) and [kc+16, kc+24), kc=(L>>4)*8
static __device__ __forceinline__ v16h load_b_frag(const _Float16* base, int lane, int ld) {
    int n  = lane & 15;
    int kc = (lane >> 4) * 8;
    const _Float16* p = base + (size_t)n * ld + kc;
    HF16 b;
    b.h[0] = *(const v8h*)(p);
    b.h[1] = *(const v8h*)(p + 16);
    return b.v;
}

// ---------------------------------------------------------------- converts
__global__ void cvt_f32_f16(const float* __restrict__ in, _Float16* __restrict__ out, int n) {
    for (int i = blockIdx.x * blockDim.x + threadIdx.x; i < n; i += gridDim.x * blockDim.x)
        out[i] = (_Float16)in[i];
}

// ------------------------------------------------- GEMM: Y = X * W^T + b
// X: [4096,2048] f16 row-major, W: [O=2048,K=2048] f16 row-major.
// Block = 8 waves: 2 along M x 4 along O; wave tile = 16 rows x 128 cols.
template<bool OUT_F32>
__global__ __launch_bounds__(256)
void gemm_xwT(const _Float16* __restrict__ A, const _Float16* __restrict__ W,
              const float* __restrict__ bias, void* __restrict__ outp) {
    const int lane  = threadIdx.x & 31;
    const int wave  = threadIdx.x >> 5;
    const int mBase = blockIdx.y * 32 + (wave & 1) * 16;
    const int oBase = blockIdx.x * 512 + (wave >> 1) * 128;

    v8f acc[8] = {};
    for (int kb = 0; kb < DIMC; kb += 32) {
        v16h a = load_a_frag(A + (size_t)mBase * DIMC + kb, lane, DIMC);
#pragma unroll
        for (int f = 0; f < 8; ++f) {
            v16h b = load_b_frag(W + (size_t)(oBase + f * 16) * DIMC + kb, lane, DIMC);
            acc[f] = __builtin_amdgcn_wmma_f32_16x16x32_f16(
                false, a, false, b, (short)0, acc[f], false, false);
        }
    }
    const int rowOff = 8 * (lane >> 4);
#pragma unroll
    for (int f = 0; f < 8; ++f) {
        int o = oBase + f * 16 + (lane & 15);
        float bsv = bias[o];
#pragma unroll
        for (int r = 0; r < 8; ++r) {
            int row = mBase + r + rowOff;
            float v = acc[f][r] + bsv;
            if (OUT_F32) ((float*)outp)[(size_t)row * DIMC + o] = v;
            else ((_Float16*)outp)[(size_t)row * DIMC + o] = (_Float16)v;
        }
    }
}

// ------------------------------------- fused RMSNorm + 3D RoPE (in place)
// One wave per (token, head). head_dim=128 -> 4 elems / lane.
__global__ __launch_bounds__(256)
void norm_rope(_Float16* __restrict__ qk, const float* __restrict__ nw, int doRope) {
    const int lane = threadIdx.x & 31;
    const int wave = threadIdx.x >> 5;
    const int p    = blockIdx.x * 8 + wave;   // (n, head) pair id
    const int n    = p >> 4;
    const int head = p & 15;
    _Float16* base = qk + (size_t)n * DIMC + head * HDIM + lane * 4;

    float x[4];
#pragma unroll
    for (int i = 0; i < 4; ++i) x[i] = (float)base[i];
    float ss = x[0]*x[0] + x[1]*x[1] + x[2]*x[2] + x[3]*x[3];
#pragma unroll
    for (int off = 1; off < 32; off <<= 1) ss += __shfl_xor(ss, off, 32);
    float inv = rsqrtf(ss * (1.0f / HDIM) + 1e-6f);
#pragma unroll
    for (int i = 0; i < 4; ++i) x[i] = x[i] * inv * nw[lane * 4 + i];

    if (doRope) {
        // n -> (t,h,w): T=8, H=16, W=32.  dt=44, dh=42, dw=42 (halves 22/21/21)
        float tf = (float)(n >> 9);
        float hf = (float)((n >> 5) & 15);
        float wf = (float)(n & 31);
#pragma unroll
        for (int pr = 0; pr < 2; ++pr) {
            int j = 2 * lane + pr;               // pair index 0..63
            float ang;
            if (j < 22)      ang = tf * __powf(10000.f, -(2.f * j)        / 44.f);
            else if (j < 43) ang = hf * __powf(10000.f, -(2.f * (j - 22)) / 42.f);
            else             ang = wf * __powf(10000.f, -(2.f * (j - 43)) / 42.f);
            float c = __cosf(ang), s = __sinf(ang);
            float e = x[2 * pr], o = x[2 * pr + 1];
            x[2 * pr]     = e * c - o * s;
            x[2 * pr + 1] = e * s + o * c;
        }
    }
#pragma unroll
    for (int i = 0; i < 4; ++i) base[i] = (_Float16)x[i];
}

// ---------------------------------------------- flash attention per head
// grid = (head=16, qBlock=32); block = 8 waves, each owns 16 query rows.
// K tiles stream via TDM (double-buffered, TENSORcnt) when available.
__global__ __launch_bounds__(256)
void flash_attn(const _Float16* __restrict__ Q, const _Float16* __restrict__ K,
                const _Float16* __restrict__ V, _Float16* __restrict__ O) {
    __shared__ _Float16 Ks[2][32][128];   // key block double buffer, row-major
    __shared__ _Float16 Vt[128][32];      // value block, transposed
    __shared__ _Float16 Ps[8][16][32];    // per-wave P staging (layout fixup)

    const int lane  = threadIdx.x & 31;
    const int wave  = threadIdx.x >> 5;
    const int head  = blockIdx.x;
    const int qBase = blockIdx.y * 128 + wave * 16;
    const size_t hoff = (size_t)head * HDIM;

    // Q fragments held in registers; fold 1/sqrt(128) into Q once (f16 mul)
    v16h qf[4];
#pragma unroll
    for (int kk = 0; kk < 4; ++kk) {
        HF16 u;
        u.v = load_a_frag(Q + (size_t)qBase * DIMC + hoff + kk * 32, lane, DIMC);
#pragma unroll
        for (int i = 0; i < 16; ++i) u.e[i] = u.e[i] * (_Float16)0.08838834764831845f;
        qf[kk] = u.v;
    }

    // all-ones B fragment: WMMA(P, ones) broadcasts row-sums into the C layout
    HF16 ob;
#pragma unroll
    for (int i = 0; i < 16; ++i) ob.e[i] = (_Float16)1.0f;
    const v16h onesB = ob.v;

    v8f acc[8] = {};
    float mRow[8], lRow[8];
#pragma unroll
    for (int r = 0; r < 8; ++r) { mRow[r] = -1e30f; lRow[r] = 0.f; }

    const int ldRow = threadIdx.x >> 3;        // 0..31 : key row
    const int ldCol = (threadIdx.x & 7) * 16;  // 0..112: head-dim chunk

#if HAVE_TDM
    if (wave == 0)   // kick off block 0 DMA before the loop (buffer 0)
        tdm_load_tile_f16(K + hoff, (unsigned int)(uintptr_t)&Ks[0][0][0]);
#endif

    for (int kb = 0; kb < NTOK; kb += 32) {
        const int buf = (kb >> 5) & 1;
#if HAVE_TDM
        if (wave == 0) {
            int nb = kb + 32;
            if (nb < NTOK) {   // issue next tile, then wait for current (in-order)
                tdm_load_tile_f16(K + (size_t)nb * DIMC + hoff,
                                  (unsigned int)(uintptr_t)&Ks[(nb >> 5) & 1][0][0]);
                WAIT_TENSOR(1);
            } else {
                WAIT_TENSOR(0);
            }
        }
#else
        {   // fallback: cooperative K staging
            const _Float16* ks = K + (size_t)(kb + ldRow) * DIMC + hoff + ldCol;
            *(v8h*)&Ks[buf][ldRow][ldCol]     = *(const v8h*)(ks);
            *(v8h*)&Ks[buf][ldRow][ldCol + 8] = *(const v8h*)(ks + 8);
        }
#endif
        {   // cooperative V staging (transposed; TDM cannot transpose)
            if (kb + 32 < NTOK)
                __builtin_prefetch(V + (size_t)(kb + 32 + ldRow) * DIMC + hoff + ldCol, 0, 3);
            const _Float16* vs = V + (size_t)(kb + ldRow) * DIMC + hoff + ldCol;
            HF16 tmp;
            tmp.h[0] = *(const v8h*)(vs);
            tmp.h[1] = *(const v8h*)(vs + 8);
#pragma unroll
            for (int i = 0; i < 16; ++i) Vt[ldCol + i][ldRow] = tmp.e[i];
        }
        __syncthreads();

        // S[16x32] = (Q*scale)(16x128) * Kblk^T : two 16x16 C tiles, 4 K-steps
        v8f s0 = {}, s1 = {};
#pragma unroll
        for (int kk = 0; kk < 4; ++kk) {
            v16h b0 = load_b_frag(&Ks[buf][0][kk * 32],  lane, 128);
            v16h b1 = load_b_frag(&Ks[buf][16][kk * 32], lane, 128);
            s0 = __builtin_amdgcn_wmma_f32_16x16x32_f16(false, qf[kk], false, b0, (short)0, s0, false, false);
            s1 = __builtin_amdgcn_wmma_f32_16x16x32_f16(false, qf[kk], false, b1, (short)0, s1, false, false);
        }

        // online softmax; row M = r + 8*(lane>>4), striped over a 16-lane DPP row
        const int rowM = 8 * (lane >> 4);
        float corrArr[8];
#pragma unroll
        for (int r = 0; r < 8; ++r) {
            float a = s0[r], b = s1[r];
            float mx = rowmax16(fmaxf(a, b));               // DPP, no LDS
            float mNew = fmaxf(mRow[r], mx);
            float corr = __expf(mRow[r] - mNew);
            corrArr[r] = corr;
            mRow[r] = mNew;
            float p0 = __expf(a - mNew);
            float p1 = __expf(b - mNew);
#pragma unroll
            for (int f = 0; f < 8; ++f) acc[f][r] *= corr;
            // stage P (C layout -> row-major in LDS) for A-fragment reload
            Ps[wave][r + rowM][lane & 15]        = (_Float16)p0;
            Ps[wave][r + rowM][16 + (lane & 15)] = (_Float16)p1;
        }

        // per-wave LDS RAW: LDS ops are in-order per wave; stop compiler reorder
        asm volatile("s_wait_dscnt 0" ::: "memory");

        v16h pf = load_a_frag(&Ps[wave][0][0], lane, 32);

        // row-sums of P via WMMA with all-ones B (replaces 32 ds_bpermute)
        v8f zsum = {};
        v8f sums = __builtin_amdgcn_wmma_f32_16x16x32_f16(false, pf, false, onesB, (short)0, zsum, false, false);
#pragma unroll
        for (int r = 0; r < 8; ++r) lRow[r] = lRow[r] * corrArr[r] + sums[r];

        // O += P(16x32) * Vblk(32x128)
#pragma unroll
        for (int f = 0; f < 8; ++f) {
            v16h vb = load_b_frag(&Vt[f * 16][0], lane, 32);
            acc[f] = __builtin_amdgcn_wmma_f32_16x16x32_f16(false, pf, false, vb, (short)0, acc[f], false, false);
        }
        __syncthreads();
    }

    const int rowM = 8 * (lane >> 4);
#pragma unroll
    for (int f = 0; f < 8; ++f) {
#pragma unroll
        for (int r = 0; r < 8; ++r) {
            int row = qBase + r + rowM;
            float v = acc[f][r] / lRow[r];
            O[(size_t)row * DIMC + hoff + f * 16 + (lane & 15)] = (_Float16)v;
        }
    }
}

// ---------------------------------------------------------------- launch
extern "C" void kernel_launch(void* const* d_in, const int* in_sizes, int n_in,
                              void* d_out, int out_size, void* d_ws, size_t ws_size,
                              hipStream_t stream) {
    (void)in_sizes; (void)n_in; (void)out_size; (void)ws_size;
    const float* x   = (const float*)d_in[0];
    const float* Wq  = (const float*)d_in[1];
    const float* bq  = (const float*)d_in[2];
    const float* Wk  = (const float*)d_in[3];
    const float* bk  = (const float*)d_in[4];
    const float* Wv  = (const float*)d_in[5];
    const float* bv  = (const float*)d_in[6];
    const float* qnw = (const float*)d_in[7];
    const float* knw = (const float*)d_in[8];
    const float* Wo  = (const float*)d_in[9];
    const float* bo  = (const float*)d_in[10];
    float* out = (float*)d_out;

    char* ws = (char*)d_ws;
    const size_t szX = (size_t)NTOK * DIMC * sizeof(_Float16);
    const size_t szW = (size_t)DIMC * DIMC * sizeof(_Float16);
    _Float16* Xh  = (_Float16*)ws; ws += szX;
    _Float16* Wqh = (_Float16*)ws; ws += szW;
    _Float16* Wkh = (_Float16*)ws; ws += szW;
    _Float16* Wvh = (_Float16*)ws; ws += szW;
    _Float16* Woh = (_Float16*)ws; ws += szW;
    _Float16* Qh  = (_Float16*)ws; ws += szX;
    _Float16* Kh  = (_Float16*)ws; ws += szX;
    _Float16* Vh  = (_Float16*)ws; ws += szX;
    _Float16* AOh = (_Float16*)ws; ws += szX;

    dim3 cblk(256), cgrid(2048);
    cvt_f32_f16<<<cgrid, cblk, 0, stream>>>(x,  Xh,  NTOK * DIMC);
    cvt_f32_f16<<<cgrid, cblk, 0, stream>>>(Wq, Wqh, DIMC * DIMC);
    cvt_f32_f16<<<cgrid, cblk, 0, stream>>>(Wk, Wkh, DIMC * DIMC);
    cvt_f32_f16<<<cgrid, cblk, 0, stream>>>(Wv, Wvh, DIMC * DIMC);
    cvt_f32_f16<<<cgrid, cblk, 0, stream>>>(Wo, Woh, DIMC * DIMC);

    dim3 ggrid(DIMC / 512, NTOK / 32), gblk(256);
    gemm_xwT<false><<<ggrid, gblk, 0, stream>>>(Xh, Wqh, bq, Qh);
    gemm_xwT<false><<<ggrid, gblk, 0, stream>>>(Xh, Wkh, bk, Kh);
    gemm_xwT<false><<<ggrid, gblk, 0, stream>>>(Xh, Wvh, bv, Vh);

    dim3 ngrid(NTOK * NHEADS / 8);
    norm_rope<<<ngrid, 256, 0, stream>>>(Qh, qnw, 1);
    norm_rope<<<ngrid, 256, 0, stream>>>(Kh, knw, 1);

    flash_attn<<<dim3(NHEADS, NTOK / 128), 256, 0, stream>>>(Qh, Kh, Vh, AOh);

    gemm_xwT<true><<<ggrid, gblk, 0, stream>>>(AOh, Woh, bo, out);
}
